// VisionTransformerCE_88983132439297
// MI455X (gfx1250) — matrix operand order repaired and checked by reading, not compiled, not run on hardware
//
#include <hip/hip_runtime.h>
#include <hip/hip_bf16.h>
#include <math.h>

// ---------------- CDNA5 WMMA plumbing ----------------
typedef _Float16 v16h __attribute__((ext_vector_type(16)));
typedef _Float16 h8   __attribute__((ext_vector_type(8)));
typedef float    v8f  __attribute__((ext_vector_type(8)));

static __device__ __forceinline__ v8f wmma_f16(v16h a, v16h b, v8f c) {
  // D = A(16x32 f16) * B(32x16 f16) + C(16x16 f32)
  return __builtin_amdgcn_wmma_f32_16x16x32_f16(
      /*neg_a=*/false, a, /*neg_b=*/false, b,
      /*c_mod=*/(short)0, c, /*reuse_a=*/false, /*reuse_b=*/false);
}

static __device__ __forceinline__ v16h cat8(h8 lo, h8 hi) {
  return __builtin_shufflevector(lo, hi, 0,1,2,3,4,5,6,7,8,9,10,11,12,13,14,15);
}

// ---------------- problem constants ----------------
#define BATCH   64
#define NTOK    352          // 16 (r) + 320 (ts) + 16 (w)
#define DIM     384
#define HEADS   6
#define HEAD_D  64
#define MLP_H   1536
#define LENS_T  64
#define LENS_S  256
#define LKEEP   180          // ceil(0.7*256)
#define NTOK2   276          // 16 + 64 + 180 + 16

// =====================================================================
// LayerNorm: one wave per token (wave32), 384 = 32 lanes * 12 elems
// =====================================================================
__global__ __launch_bounds__(256) void ln_kernel(
    const float* __restrict__ X, const float* __restrict__ w,
    const float* __restrict__ bb, float* __restrict__ Y, int nTok)
{
  const int wave = threadIdx.x >> 5;
  const int lane = threadIdx.x & 31;
  const int tok  = blockIdx.x * 8 + wave;
  if (tok >= nTok) return;
  const float* row = X + (size_t)tok * DIM;
  float v[12];
  float s = 0.f;
#pragma unroll
  for (int i = 0; i < 12; ++i) { v[i] = row[lane + i * 32]; s += v[i]; }
#pragma unroll
  for (int off = 16; off; off >>= 1) s += __shfl_xor(s, off, 32);
  const float mean = s * (1.f / DIM);
  float var = 0.f;
#pragma unroll
  for (int i = 0; i < 12; ++i) { float d = v[i] - mean; var += d * d; }
#pragma unroll
  for (int off = 16; off; off >>= 1) var += __shfl_xor(var, off, 32);
  const float rs = rsqrtf(var * (1.f / DIM) + 1e-5f);
  float* out = Y + (size_t)tok * DIM;
#pragma unroll
  for (int i = 0; i < 12; ++i) {
    const int c = lane + i * 32;
    out[c] = (v[i] - mean) * rs * w[c] + bb[c];
  }
}

// =====================================================================
// Tiled WMMA GEMM: C[M,N] = A[M,K] * B[K,N] (+bias, +residual / +GELU)
// 256 threads = 8 waves; 64x64 block tile; BK=32; A,B^T staged in LDS f16.
// M % 64 == 0 and N % 64 == 0 for every call site (no edge predication,
// EXEC stays all-ones around WMMA as required).
// EPI: 0 = bias; 1 = bias+residual; 2 = bias+GELU(exact erf)
// =====================================================================
template <typename TA, typename TO, int EPI>
__global__ __launch_bounds__(256) void gemm_wmma(
    const TA* __restrict__ A, const float* __restrict__ Bw,
    const float* __restrict__ bias, const float* __restrict__ res,
    TO* __restrict__ Out, int M, int N, int K)
{
  __shared__ __align__(16) _Float16 As[64][40];   // [m][k], pad->80B stride
  __shared__ __align__(16) _Float16 Bst[64][40];  // [n][k] (transposed)

  const int tid  = threadIdx.x;
  const int lane = tid & 31;
  const int w    = tid >> 5;        // wave 0..7
  const int mi   = w & 3;           // 16-row tile within block
  const int nh   = w >> 2;          // 32-col half within block
  const int rowBase = blockIdx.y * 64;
  const int colBase = blockIdx.x * 64;

  const int aRow = tid >> 2, aSeg = (tid & 3) * 8;   // A: 64 rows x 32 cols
  const int bK   = tid >> 3, bSeg = (tid & 7) * 8;   // B: 32 rows x 64 cols

  v8f acc[2] = {};

  for (int kt = 0; kt < K; kt += 32) {
    // ---- stage A (convert to f16) ----
    {
      const TA* src = A + (size_t)(rowBase + aRow) * K + kt + aSeg;
      if (kt + 32 < K) __builtin_prefetch(src + 32, 0, 1);
      h8 hv;
      if constexpr (sizeof(TA) == 4) {
        const float4 f0 = *(const float4*)(src);
        const float4 f1 = *(const float4*)(src + 4);
        hv[0]=(_Float16)f0.x; hv[1]=(_Float16)f0.y; hv[2]=(_Float16)f0.z; hv[3]=(_Float16)f0.w;
        hv[4]=(_Float16)f1.x; hv[5]=(_Float16)f1.y; hv[6]=(_Float16)f1.z; hv[7]=(_Float16)f1.w;
      } else {
        hv = *(const h8*)(src);
      }
      *(h8*)&As[aRow][aSeg] = hv;
    }
    // ---- stage B transposed (convert to f16) ----
    {
      const float* src = Bw + (size_t)(kt + bK) * N + colBase + bSeg;
      const float4 f0 = *(const float4*)(src);
      const float4 f1 = *(const float4*)(src + 4);
      Bst[bSeg + 0][bK] = (_Float16)f0.x;
      Bst[bSeg + 1][bK] = (_Float16)f0.y;
      Bst[bSeg + 2][bK] = (_Float16)f0.z;
      Bst[bSeg + 3][bK] = (_Float16)f0.w;
      Bst[bSeg + 4][bK] = (_Float16)f1.x;
      Bst[bSeg + 5][bK] = (_Float16)f1.y;
      Bst[bSeg + 6][bK] = (_Float16)f1.z;
      Bst[bSeg + 7][bK] = (_Float16)f1.w;
    }
    __syncthreads();

    // ---- A fragment (16x32): lane holds row M=lane&15; half runs per ISA layout ----
    const int ar = mi * 16 + (lane & 15);
    const int kShift = (lane >> 4) * 8;
    const v16h afrag = cat8(*(const h8*)&As[ar][kShift],
                            *(const h8*)&As[ar][16 + kShift]);

    // ---- two B fragments (32x16): lane holds col N=lane&15; half runs per ISA layout ----
    const int koff = (lane >> 4) * 8;
#pragma unroll
    for (int p = 0; p < 2; ++p) {
      const int n = (nh * 2 + p) * 16 + (lane & 15);
      const v16h bfrag = cat8(*(const h8*)&Bst[n][koff],
                              *(const h8*)&Bst[n][koff + 16]);
      acc[p] = wmma_f16(afrag, bfrag, acc[p]);
    }
    __syncthreads();
  }

  // ---- epilogue: D layout -> lane holds N=lane&15, vgpr r holds M=r+(lane>=16)*8
  const int laneN = lane & 15, hiG = lane >> 4;
#pragma unroll
  for (int p = 0; p < 2; ++p) {
    const int gn = colBase + (nh * 2 + p) * 16 + laneN;
    const float bv = bias[gn];
#pragma unroll
    for (int r = 0; r < 8; ++r) {
      const int gm = rowBase + mi * 16 + r + hiG * 8;
      float vv = acc[p][r] + bv;
      if constexpr (EPI == 1) vv += res[(size_t)gm * N + gn];
      if constexpr (EPI == 2) vv = 0.5f * vv * (1.f + erff(vv * 0.70710678118654752f));
      if constexpr (sizeof(TO) == 2) Out[(size_t)gm * N + gn] = (_Float16)vv;
      else                           Out[(size_t)gm * N + gn] = vv;
    }
  }
}

// =====================================================================
// Attention: one wave per (batch, head, 16-query tile).
// Segments: qt==0 -> r (Lk=16); qt 1..20 -> ts (Lk=336); qt==21 -> w (Lk=352).
// S = softmax(Q K^T * 0.125) kept in LDS; O = S V. All GEMM work via WMMA.
// Template-query tiles (qt 1..4) accumulate search-token attention mass.
// =====================================================================
__global__ __launch_bounds__(32) void attn_kernel(
    const _Float16* __restrict__ qkv,   // [B*NTOK, 1152] f16 (q|k|v per head)
    float* __restrict__ attnOut,        // [B*NTOK, 384] f32
    float* __restrict__ attnT)          // [B, 256] accum
{
  __shared__ __align__(16) float S[16][NTOK];

  const int lane = threadIdx.x;
  const int qt = blockIdx.x, hh = blockIdx.y, b = blockIdx.z;
  int qbase, Lk;
  if (qt == 0)       { qbase = 0;                  Lk = 16;  }
  else if (qt <= 20) { qbase = 16 + (qt - 1) * 16; Lk = 336; }
  else               { qbase = 336;                Lk = 352; }

  const int laneN = lane & 15, hiHalf = lane >> 4;

  // ---- Q fragments (row M = qbase+laneN), two 32-wide K steps over HEAD_D=64
  const _Float16* qp = qkv + ((size_t)(b * NTOK + qbase + laneN)) * 1152 + hh * HEAD_D;
  v16h qf[2];
#pragma unroll
  for (int ds = 0; ds < 2; ++ds) {
    const _Float16* p = qp + ds * 32 + hiHalf * 8;
    qf[ds] = cat8(*(const h8*)p, *(const h8*)(p + 16));
  }

  // ---- scores: S[16][Lk], 16-key tiles (Lk is a multiple of 16 in all cases)
  const int nkt = Lk >> 4;
  for (int kt = 0; kt < nkt; ++kt) {
    const int ktok = kt * 16 + laneN;   // B-fragment column = key token
    const _Float16* kp = qkv + ((size_t)(b * NTOK + ktok)) * 1152
                         + DIM + hh * HEAD_D + hiHalf * 8;
    v8f c = {};
#pragma unroll
    for (int ds = 0; ds < 2; ++ds) {
      const v16h bf = cat8(*(const h8*)(kp + ds * 32),
                           *(const h8*)(kp + ds * 32 + 16));
      c = wmma_f16(qf[ds], bf, c);
    }
#pragma unroll
    for (int r = 0; r < 8; ++r)
      S[r + hiHalf * 8][kt * 16 + laneN] = c[r] * 0.125f;   // 64^-0.5
  }

  // ---- row softmax (lanes 0..15 own one row each); single wave -> LDS in-order
  if (lane < 16) {
    float m = -1e30f;
    for (int k = 0; k < Lk; ++k) m = fmaxf(m, S[lane][k]);
    float s = 0.f;
    for (int k = 0; k < Lk; ++k) { const float e = __expf(S[lane][k] - m); S[lane][k] = e; s += e; }
    const float inv = 1.f / s;
    for (int k = 0; k < Lk; ++k) S[lane][k] *= inv;
    // template queries (global tokens 16..79) feed search-token pruning scores
    if (qt >= 1 && qt <= 4) {
      for (int s2 = 0; s2 < LENS_S; ++s2)
        atomicAdd(&attnT[b * LENS_S + s2], S[lane][80 + s2]);
    }
  }
  // reconverged: EXEC all-ones again for the WMMAs below

  // ---- O = P * V : 4 column tiles of HEAD_D, K steps of 32 (last ts step padded)
  v8f o[4] = {};
  const int nk32 = (Lk + 31) >> 5;
  for (int ks = 0; ks < nk32; ++ks) {
    v16h pf;
#pragma unroll
    for (int h = 0; h < 16; ++h) {
      const int kk = ks * 32 + (h < 8 ? h : h + 8) + hiHalf * 8;
      pf[h] = (kk < Lk) ? (_Float16)S[laneN][kk] : (_Float16)0.f;
    }
#pragma unroll
    for (int ni = 0; ni < 4; ++ni) {
      v16h vf;
#pragma unroll
      for (int h = 0; h < 16; ++h) {
        const int ktok = ks * 32 + (h < 8 ? h : h + 8) + hiHalf * 8;     // < NTOK always
        vf[h] = qkv[((size_t)(b * NTOK + ktok)) * 1152
                    + 2 * DIM + hh * HEAD_D + ni * 16 + laneN];
      }
      o[ni] = wmma_f16(pf, vf, o[ni]);
    }
  }
#pragma unroll
  for (int ni = 0; ni < 4; ++ni)
#pragma unroll
    for (int r = 0; r < 8; ++r) {
      const int q = qbase + r + hiHalf * 8;
      attnOut[((size_t)(b * NTOK + q)) * DIM + hh * HEAD_D + ni * 16 + laneN] = o[ni][r];
    }
}

// =====================================================================
// Per-batch stable top-k: bitonic sort 256 (value desc, index asc ties)
// == jnp.argsort(-attn_t)[:, :180]
// =====================================================================
__global__ __launch_bounds__(256) void topk_kernel(
    const float* __restrict__ attnT, int* __restrict__ topk)
{
  __shared__ float sv[256];
  __shared__ int   si[256];
  const int t = threadIdx.x, b = blockIdx.x;
  sv[t] = attnT[b * LENS_S + t];
  si[t] = t;
  __syncthreads();
  for (int k = 2; k <= 256; k <<= 1)
    for (int j = k >> 1; j > 0; j >>= 1) {
      const int ixj = t ^ j;
      if (ixj > t) {
        const float v0 = sv[t], v1 = sv[ixj];
        const int   i0 = si[t], i1 = si[ixj];
        const bool before = (v0 > v1) || (v0 == v1 && i0 < i1); // desired order
        const bool asc = ((t & k) == 0);
        if (asc ? !before : before) {
          sv[t] = v1; sv[ixj] = v0; si[t] = i1; si[ixj] = i0;
        }
      }
      __syncthreads();
    }
  if (t < LKEEP) topk[b * LKEEP + t] = si[t];
}

// =====================================================================
// Gather pruned token set: [r(16) | t(64) | s_kept(180) | w(16)] -> 276 tokens
// =====================================================================
__global__ __launch_bounds__(256) void gather_kernel(
    const float* __restrict__ xr, const int* __restrict__ topk,
    float* __restrict__ x2)
{
  const int i = blockIdx.x * 256 + threadIdx.x;   // grid sized exactly
  const int c = i % DIM;
  const int t = (i / DIM) % NTOK2;
  const int b = i / (DIM * NTOK2);
  int src;
  if (t < 80)       src = t;                                  // r + template
  else if (t < 260) src = 80 + topk[b * LKEEP + (t - 80)];    // kept search
  else              src = 336 + (t - 260);                    // w tokens
  x2[i] = xr[((size_t)(b * NTOK + src)) * DIM + c];
}

__global__ __launch_bounds__(256) void zero_kernel(float* p, int n) {
  const int i = blockIdx.x * 256 + threadIdx.x;
  if (i < n) p[i] = 0.f;
}

// =====================================================================
// host-side orchestration
// =====================================================================
extern "C" void kernel_launch(void* const* d_in, const int* in_sizes, int n_in,
                              void* d_out, int out_size, void* d_ws, size_t ws_size,
                              hipStream_t stream) {
  (void)in_sizes; (void)n_in; (void)out_size; (void)ws_size;
  const float* x       = (const float*)d_in[0];
  // d_in[1], d_in[2]: arange index templates (unused by the math)
  const float* norm1_w = (const float*)d_in[3];
  const float* norm1_b = (const float*)d_in[4];
  const float* qkv_w   = (const float*)d_in[5];
  const float* qkv_b   = (const float*)d_in[6];
  const float* proj_w  = (const float*)d_in[7];
  const float* proj_b  = (const float*)d_in[8];
  const float* norm2_w = (const float*)d_in[9];
  const float* norm2_b = (const float*)d_in[10];
  const float* fc1_w   = (const float*)d_in[11];
  const float* fc1_b   = (const float*)d_in[12];
  const float* fc2_w   = (const float*)d_in[13];
  const float* fc2_b   = (const float*)d_in[14];
  float* out = (float*)d_out;

  const int M1 = BATCH * NTOK;    // 22528
  const int M2 = BATCH * NTOK2;   // 17664

  // workspace layout (bytes, all 256-aligned); reuse across phases
  char* ws = (char*)d_ws;
  const size_t SZ_F16BIG = 54263808;          // max(qkv f16, fc1 f16)
  const size_t SZ_F32BIG = (size_t)M1 * DIM * 4;  // 34603008
  _Float16* qkv_h  = (_Float16*)(ws);                           // then fc1_h
  float*    xn     = (float*)(ws + SZ_F16BIG);                  // then hln
  float*    attnO  = (float*)(ws + SZ_F16BIG + SZ_F32BIG);      // then x2
  float*    x_res  = (float*)(ws + SZ_F16BIG + 2 * SZ_F32BIG);
  float*    attnT  = (float*)(ws + SZ_F16BIG + 3 * SZ_F32BIG);
  int*      topk   = (int*)  (ws + SZ_F16BIG + 3 * SZ_F32BIG + 65536);
  _Float16* fc1_h  = qkv_h;
  float*    hln    = xn;
  float*    x2     = attnO;

  // 1) LN1
  ln_kernel<<<M1 / 8, 256, 0, stream>>>(x, norm1_w, norm1_b, xn, M1);
  // 2) QKV GEMM  (22528 x 1152 x 384), f16 out
  gemm_wmma<float, _Float16, 0><<<dim3(1152 / 64, M1 / 64), 256, 0, stream>>>(
      xn, qkv_w, qkv_b, nullptr, qkv_h, M1, 1152, DIM);
  // 3) clear pruning-score accumulator
  zero_kernel<<<(BATCH * LENS_S) / 256, 256, 0, stream>>>(attnT, BATCH * LENS_S);
  // 4) segmented attention (r / ts / w), one wave per 16-query tile
  attn_kernel<<<dim3(22, HEADS, BATCH), 32, 0, stream>>>(qkv_h, attnO, attnT);
  // 5) proj GEMM fused with residual: x_res = x + attnO @ proj_w + proj_b
  gemm_wmma<float, float, 1><<<dim3(DIM / 64, M1 / 64), 256, 0, stream>>>(
      attnO, proj_w, proj_b, x, x_res, M1, DIM, DIM);
  // 6) stable top-180 of search tokens per batch
  topk_kernel<<<BATCH, 256, 0, stream>>>(attnT, topk);
  // 7) gather pruned token set -> x2 [B, 276, 384]
  gather_kernel<<<(BATCH * NTOK2 * DIM) / 256, 256, 0, stream>>>(x_res, topk, x2);
  // 8) LN2
  ln_kernel<<<M2 / 8, 256, 0, stream>>>(x2, norm2_w, norm2_b, hln, M2);
  // 9) FC1 GEMM fused with exact GELU, f16 out (17664 x 1536 x 384)
  gemm_wmma<float, _Float16, 2><<<dim3(MLP_H / 64, M2 / 64), 256, 0, stream>>>(
      hln, fc1_w, fc1_b, nullptr, fc1_h, M2, MLP_H, DIM);
  // 10) FC2 GEMM fused with bias + residual -> d_out (17664 x 384 x 1536)
  gemm_wmma<_Float16, float, 1><<<dim3(DIM / 64, M2 / 64), 256, 0, stream>>>(
      fc1_h, fc2_w, fc2_b, x2, out, M2, DIM, MLP_H);
}
